// GCNN_3p_81063212744716
// MI455X (gfx1250) — matrix operand order repaired and checked by reading, not compiled, not run on hardware
//
#include <hip/hip_runtime.h>
#include <hip/hip_bf16.h>
#include <math.h>

// ---------------- problem constants (match reference) ----------------
#define NN   50000
#define EE   800000
#define CC   64
#define BB   128
#define OUTF 2
#define EPSV 1e-5f

typedef __attribute__((ext_vector_type(16))) _Float16 v16h;
typedef __attribute__((ext_vector_type(8)))  float    v8f;

// ---------------- gcn_norm ----------------
__global__ void gcn_deg_init(float* __restrict__ deg) {
    int i = blockIdx.x * 256 + threadIdx.x;
    if (i < NN) deg[i] = 1.0f;              // self-loop weight folded in
}

__global__ void gcn_deg_edges(const int* __restrict__ col,
                              const float* __restrict__ w,
                              float* __restrict__ deg) {
    int e = blockIdx.x * 256 + threadIdx.x;
    if (e < EE) atomicAdd(&deg[col[e]], w[e]);
}

__global__ void gcn_dinv(float* __restrict__ deg) {
    int i = blockIdx.x * 256 + threadIdx.x;
    if (i < NN) {
        float d = deg[i];
        deg[i] = (d > 0.0f) ? rsqrtf(d) : 0.0f;   // in place: deg -> dinv
    }
}

__global__ void gcn_norm(const int* __restrict__ row,
                         const int* __restrict__ col,
                         const float* __restrict__ w,
                         const float* __restrict__ dinv,
                         float* __restrict__ normE,
                         float* __restrict__ normS) {
    int t = blockIdx.x * 256 + threadIdx.x;
    if (t < EE) normE[t] = dinv[row[t]] * w[t] * dinv[col[t]];
    if (t < NN) normS[t] = dinv[t] * dinv[t];
}

__global__ void gcn_zero(float* __restrict__ p, int n) {
    int i = blockIdx.x * 256 + threadIdx.x;
    if (i < n) p[i] = 0.0f;
}

// ---------------- dense GEMM  H = X(N,64) @ W(64,64)  via v_wmma_f32_16x16x32_f16 ----------------
// Block = 256 threads = 8 waves; each wave computes one 16-row strip across all 64 cols.
__global__ void gcn_gemm64_wmma(const float* __restrict__ X,
                                const float* __restrict__ W,
                                float* __restrict__ Hout,
                                int nRows) {
    __shared__ __align__(32) _Float16 WT[64 * 64];   // transposed: WT[n*64 + k]
    const int tid = threadIdx.x;
    for (int idx = tid; idx < 64 * 64; idx += 256) {
        int k = idx >> 6, n = idx & 63;
        WT[n * 64 + k] = (_Float16)W[idx];           // W row-major [k][n]
    }
    __syncthreads();

    const int wave  = tid >> 5;
    const int lane  = tid & 31;
    const int strip = blockIdx.x * 8 + wave;
    if (strip * 16 >= nRows) return;                 // wave-uniform exit (EXEC stays all-1)

    const int m  = lane & 15;                        // A row within tile (both lane halves)
    const int kh = lane >> 4;                        // K-half selector
    const int n  = lane & 15;                        // B/D column within tile
    const size_t rowBase = (size_t)(strip * 16 + m) * 64;

    v8f c0 = {}, c1 = {}, c2 = {}, c3 = {};

    #pragma unroll
    for (int ks = 0; ks < 2; ++ks) {
        // A fragment: contiguous k order (k = 16*kh + element), same as B.
        // lanes 0-15:  v0..7 -> K=2v..2v+1; lanes 16-31: +16
        v16h a;
        #pragma unroll
        for (int v = 0; v < 8; ++v) {
            int kb = 2 * v + 16 * kh + ks * 32;
            a[2 * v]     = (_Float16)X[rowBase + kb];
            a[2 * v + 1] = (_Float16)X[rowBase + kb + 1];
        }
        // B fragments: lane(khalf,n); half h -> K = ks*32 + kh*16 + h; contiguous in WT.
        const v16h b0 = *(const v16h*)&WT[(0 * 16 + n) * 64 + ks * 32 + kh * 16];
        const v16h b1 = *(const v16h*)&WT[(1 * 16 + n) * 64 + ks * 32 + kh * 16];
        const v16h b2 = *(const v16h*)&WT[(2 * 16 + n) * 64 + ks * 32 + kh * 16];
        const v16h b3 = *(const v16h*)&WT[(3 * 16 + n) * 64 + ks * 32 + kh * 16];

        c0 = __builtin_amdgcn_wmma_f32_16x16x32_f16(false, a, false, b0, (short)0, c0, false, false);
        c1 = __builtin_amdgcn_wmma_f32_16x16x32_f16(false, a, false, b1, (short)0, c1, false, false);
        c2 = __builtin_amdgcn_wmma_f32_16x16x32_f16(false, a, false, b2, (short)0, c2, false, false);
        c3 = __builtin_amdgcn_wmma_f32_16x16x32_f16(false, a, false, b3, (short)0, c3, false, false);
    }

    // D layout: VGPR r -> M = r + 8*kh, N = lane&15
    #pragma unroll
    for (int r = 0; r < 8; ++r) {
        const int mr = r + 8 * kh;
        const size_t base = (size_t)(strip * 16 + mr) * 64 + n;
        Hout[base +  0] = c0[r];
        Hout[base + 16] = c1[r];
        Hout[base + 32] = c2[r];
        Hout[base + 48] = c3[r];
    }
}

// ---------------- edge scatter:  agg[col] += h[row] * norm ----------------
__global__ void gcn_aggregate(const float* __restrict__ H,
                              const int* __restrict__ row,
                              const int* __restrict__ col,
                              const float* __restrict__ normE,
                              float* __restrict__ agg) {
    long long t = (long long)blockIdx.x * 256 + threadIdx.x;
    if (t >= (long long)EE * 16) return;
    const int e = (int)(t >> 4);
    const int q = ((int)t & 15) * 4;
    const int r = row[e], c = col[e];
    const float nm = normE[e];
    const float4 v = *(const float4*)&H[(size_t)r * 64 + q];
    float* dst = &agg[(size_t)c * 64 + q];
    atomicAdd(dst + 0, v.x * nm);
    atomicAdd(dst + 1, v.y * nm);
    atomicAdd(dst + 2, v.z * nm);
    atomicAdd(dst + 3, v.w * nm);
}

// ---------------- self-loop + bias + ReLU + BN (in place on agg) ----------------
__global__ void gcn_finalize(float* __restrict__ agg,
                             const float* __restrict__ H,
                             const float* __restrict__ normS,
                             const float* __restrict__ bias,
                             const float* __restrict__ gamma,
                             const float* __restrict__ beta,
                             const float* __restrict__ mean,
                             const float* __restrict__ var) {
    int t = blockIdx.x * 256 + threadIdx.x;
    if (t >= NN * 64) return;
    const int i = t >> 6, ch = t & 63;
    float v = agg[t] + H[t] * normS[i] + bias[ch];
    v = fmaxf(v, 0.0f);
    v = (v - mean[ch]) * rsqrtf(var[ch] + EPSV) * gamma[ch] + beta[ch];
    agg[t] = v;
}

// ---------------- pooling ----------------
__device__ inline void atomicMaxF(float* addr, float val) {
    int* ia = (int*)addr;
    int old = __float_as_int(*addr);
    while (__int_as_float(old) < val) {
        int assumed = old;
        old = atomicCAS(ia, assumed, __float_as_int(val));
        if (old == assumed) break;
    }
}

__global__ void gcn_pool_init(float* __restrict__ psum,
                              float* __restrict__ pmax,
                              float* __restrict__ pcnt) {
    int t = blockIdx.x * 256 + threadIdx.x;
    if (t < BB * 64) { psum[t] = 0.0f; pmax[t] = -__builtin_huge_valf(); }
    if (t < BB) pcnt[t] = 0.0f;
}

__global__ void gcn_pool(const float* __restrict__ H,
                         const int* __restrict__ batch,
                         float* __restrict__ psum,
                         float* __restrict__ pmax,
                         float* __restrict__ pcnt) {
    int t = blockIdx.x * 256 + threadIdx.x;
    if (t >= NN * 64) return;
    const int i = t >> 6, ch = t & 63;
    const int b = batch[i];
    const float v = H[t];
    atomicAdd(&psum[b * 64 + ch], v);
    atomicMaxF(&pmax[b * 64 + ch], v);
    if (ch == 0) atomicAdd(&pcnt[b], 1.0f);
}

// ---------------- head: concat(sum, mean, max) @ lin_w + lin_b ----------------
__global__ void gcn_head(const float* __restrict__ psum,
                         const float* __restrict__ pmax,
                         const float* __restrict__ pcnt,
                         const float* __restrict__ lw,   // (192, 2) row-major
                         const float* __restrict__ lb,
                         float* __restrict__ out) {
    int t = threadIdx.x;                 // 256 = 128 graphs * 2 outputs
    const int b = t >> 1, o = t & 1;
    const float cn = fmaxf(pcnt[b], 1.0f);
    float acc = lb[o];
    for (int j = 0; j < 64; ++j) {
        const float s = psum[b * 64 + j];
        float mx = pmax[b * 64 + j];
        if (mx == -__builtin_huge_valf()) mx = 0.0f;   // jnp.where(isneginf, 0, ...)
        acc += s         * lw[(j)       * 2 + o];
        acc += (s / cn)  * lw[(64 + j)  * 2 + o];
        acc += mx        * lw[(128 + j) * 2 + o];
    }
    out[b * 2 + o] = acc;
}

// ---------------- launch ----------------
extern "C" void kernel_launch(void* const* d_in, const int* in_sizes, int n_in,
                              void* d_out, int out_size, void* d_ws, size_t ws_size,
                              hipStream_t stream) {
    (void)in_sizes; (void)n_in; (void)out_size; (void)ws_size;

    const float* x       = (const float*)d_in[0];
    const float* ew      = (const float*)d_in[1];
    const float* conv1_w = (const float*)d_in[2];
    const float* conv1_b = (const float*)d_in[3];
    const float* conv2_w = (const float*)d_in[4];
    const float* conv2_b = (const float*)d_in[5];
    const float* bn1_g   = (const float*)d_in[6];
    const float* bn1_b   = (const float*)d_in[7];
    const float* bn1_m   = (const float*)d_in[8];
    const float* bn1_v   = (const float*)d_in[9];
    const float* bn2_g   = (const float*)d_in[10];
    const float* bn2_b   = (const float*)d_in[11];
    const float* bn2_m   = (const float*)d_in[12];
    const float* bn2_v   = (const float*)d_in[13];
    const float* lin_w   = (const float*)d_in[14];
    const float* lin_b   = (const float*)d_in[15];
    const int*   eidx    = (const int*)d_in[16];
    const int*   batch   = (const int*)d_in[17];
    const int*   erow    = eidx;          // edge_index[0]
    const int*   ecol    = eidx + EE;     // edge_index[1]
    float*       out     = (float*)d_out;

    // workspace carve-out (floats): ~29.3 MB total
    float* ws    = (float*)d_ws;
    float* deg   = ws;                               // N   (becomes dinv in place)
    float* normE = deg   + NN;                       // E
    float* normS = normE + EE;                       // N
    float* bufH  = normS + NN;                       // N*64  (GEMM output h)
    float* bufA  = bufH  + (size_t)NN * 64;          // N*64  (aggregate / bn output)
    float* psum  = bufA  + (size_t)NN * 64;          // B*64
    float* pmax  = psum  + BB * 64;                  // B*64
    float* pcnt  = pmax  + BB * 64;                  // B

    const int gN   = (NN + 255) / 256;
    const int gE   = (EE + 255) / 256;
    const int gNE  = ((EE > NN ? EE : NN) + 255) / 256;
    const int gNC  = (NN * 64 + 255) / 256;
    const int gAgg = (int)(((long long)EE * 16 + 255) / 256);
    const int gGemm = ((NN / 16) + 7) / 8;           // 3125 strips / 8 waves per block

    // gcn_norm
    gcn_deg_init <<<gN,  256, 0, stream>>>(deg);
    gcn_deg_edges<<<gE,  256, 0, stream>>>(ecol, ew, deg);
    gcn_dinv     <<<gN,  256, 0, stream>>>(deg);
    gcn_norm     <<<gNE, 256, 0, stream>>>(erow, ecol, ew, deg, normE, normS);

    // ---- layer 1 ----
    gcn_gemm64_wmma<<<gGemm, 256, 0, stream>>>(x, conv1_w, bufH, NN);
    gcn_zero       <<<gNC,   256, 0, stream>>>(bufA, NN * 64);
    gcn_aggregate  <<<gAgg,  256, 0, stream>>>(bufH, erow, ecol, normE, bufA);
    gcn_finalize   <<<gNC,   256, 0, stream>>>(bufA, bufH, normS, conv1_b,
                                               bn1_g, bn1_b, bn1_m, bn1_v);

    // ---- layer 2 (bufA holds h1_bn; GEMM2 finishes before bufA is re-zeroed) ----
    gcn_gemm64_wmma<<<gGemm, 256, 0, stream>>>(bufA, conv2_w, bufH, NN);
    gcn_zero       <<<gNC,   256, 0, stream>>>(bufA, NN * 64);
    gcn_aggregate  <<<gAgg,  256, 0, stream>>>(bufH, erow, ecol, normE, bufA);
    gcn_finalize   <<<gNC,   256, 0, stream>>>(bufA, bufH, normS, conv2_b,
                                               bn2_g, bn2_b, bn2_m, bn2_v);

    // ---- pooling + head ----
    gcn_pool_init<<<(BB * 64 + 255) / 256, 256, 0, stream>>>(psum, pmax, pcnt);
    gcn_pool     <<<gNC, 256, 0, stream>>>(bufA, batch, psum, pmax, pcnt);
    gcn_head     <<<1, BB * OUTF, 0, stream>>>(psum, pmax, pcnt, lin_w, lin_b, out);
}
